// HADES_29532195127575
// MI455X (gfx1250) — compile-verified, matrix operand order checked by reading, not run on hardware
//
#include <hip/hip_runtime.h>
#include <hip/hip_bf16.h>

#define DEV __device__ __forceinline__

typedef __attribute__((ext_vector_type(16))) __bf16 v16bf;
typedef __attribute__((ext_vector_type(8)))  __bf16 v8bf;
typedef __attribute__((ext_vector_type(8)))  float  v8f;

// ---- problem sizes (match reference) ----
constexpr int B_     = 4;
constexpr int L_     = 2048;
constexpr int DM     = 1024;   // D_MODEL
constexpr int BL     = B_ * L_;        // 8192 rows
constexpr int NPROJ  = 2336;   // D_IN_PROJ
constexpr int DSSM   = 1024;   // D_INNER == D_SSM
constexpr int CONVD  = 1280;   // CONV_DIM
constexpr int DSTATE = 128;
constexpr int NH     = 16;     // NUM_FILTERS
constexpr int HD     = 64;     // HEADDIM
constexpr int HIN    = 1056;   // H_IN
constexpr int HOUT   = 40;     // H_OUT
constexpr int NSEL   = 12;     // N_SELECT
constexpr int SELF   = 28;     // SELECT_FILTERS
constexpr float EPSV = 1e-5f;

// =====================================================================
// f32 -> bf16 conversion (grid-stride)
// =====================================================================
__global__ void cvt_bf16_kernel(const float* __restrict__ src,
                                __bf16* __restrict__ dst, int n) {
  for (int i = blockIdx.x * blockDim.x + threadIdx.x; i < n;
       i += gridDim.x * blockDim.x)
    dst[i] = (__bf16)src[i];
}

// =====================================================================
// bf16 WMMA GEMM: C[M,N] f32 = A[M,K] (bf16 rm) * W[N,K]^T (bf16 rm)
// wave tile = 32(M) x 32(N): 4 fragment loads -> 4 WMMAs per K-step
// (1:1 vmem:wmma). 4 waves/block along M (block tile 128M x 32N).
// Fragment layout per CDNA5 ISA 7.12.2 (16-bit A 16x32).
// =====================================================================
DEV v16bf load_frag(const __bf16* __restrict__ p) {
  v8bf lo = *(const v8bf*)(p);
  v8bf hi = *(const v8bf*)(p + 16);
  v16bf f;
#pragma unroll
  for (int i = 0; i < 8; ++i) { f[i] = lo[i]; f[i + 8] = hi[i]; }
  return f;
}

DEV v8f wmma_bf16(v16bf a, v16bf b, v8f c) {
  return __builtin_amdgcn_wmma_f32_16x16x32_bf16(false, a, false, b,
                                                 (short)0, c, false, false);
}

__global__ __launch_bounds__(128) void gemm_bf16_wmma(
    const __bf16* __restrict__ A, const __bf16* __restrict__ W,
    float* __restrict__ C, int M, int N, int K) {
  const int lane = threadIdx.x & 31;
  const int wave = threadIdx.x >> 5;
  const int m0 = (blockIdx.y * 4 + wave) * 32;
  const int n0 = blockIdx.x * 32;
  const int rsel = lane & 15;
  const int koff = (lane >> 4) * 8;   // K phase per half-wave

  const __bf16* pa0 = A + (long)(m0 + rsel) * K + koff;
  const __bf16* pa1 = A + (long)(m0 + 16 + rsel) * K + koff;
  const __bf16* pb0 = W + (long)(n0 + rsel) * K + koff;
  const __bf16* pb1 = W + (long)(n0 + 16 + rsel) * K + koff;

  v8f acc00 = {}, acc01 = {}, acc10 = {}, acc11 = {};
  for (int k0 = 0; k0 < K; k0 += 32) {
    // pull the K-tile ~512B ahead toward the WGP (global_prefetch_b8;
    // speculative -> OOB at the tail is silently dropped)
    __builtin_prefetch(pa0 + k0 + 256, 0, 3);
    __builtin_prefetch(pb0 + k0 + 256, 0, 3);
    v16bf a0 = load_frag(pa0 + k0);
    v16bf a1 = load_frag(pa1 + k0);
    v16bf b0 = load_frag(pb0 + k0);
    v16bf b1 = load_frag(pb1 + k0);
    acc00 = wmma_bf16(a0, b0, acc00);
    acc01 = wmma_bf16(a0, b1, acc01);
    acc10 = wmma_bf16(a1, b0, acc10);
    acc11 = wmma_bf16(a1, b1, acc11);
  }
  // C/D layout: VGPR r -> lanes 0-15: (M=r, N=lane); lanes 16-31: (M=8+r)
  const int col = lane & 15;
  const int mr  = (lane >> 4) * 8;
  float* c0 = C + (long)(m0 + mr) * N + n0 + col;
  float* c1 = C + (long)(m0 + 16 + mr) * N + n0 + col;
#pragma unroll
  for (int r = 0; r < 8; ++r) {
    c0[(long)r * N]      = acc00[r];
    c0[(long)r * N + 16] = acc01[r];
    c1[(long)r * N]      = acc10[r];
    c1[(long)r * N + 16] = acc11[r];
  }
}

// =====================================================================
// spectral residual: sp[b,t,d] = u[b,t,d] - cumsum_t(u)/(t+1)
// one thread per (b,d); coalesced in d.
// =====================================================================
__global__ void spectral_kernel(const float* __restrict__ u,
                                float* __restrict__ sp) {
  const int b = blockIdx.x >> 2;
  const int d = ((blockIdx.x & 3) << 8) + threadIdx.x;
  const float* pu = u + (long)b * L_ * DM + d;
  float* po = sp + (long)b * L_ * DM + d;
  float acc = 0.f;
  for (int t = 0; t < L_; ++t) {
    float v = pu[(long)t * DM];
    acc += v;
    po[(long)t * DM] = v - acc / (float)(t + 1);
  }
}

// =====================================================================
// h_proj (N=40, VALU) + top-k(28->12) + dt gather + softplus + decay
// one block per (b,l) row.
// =====================================================================
__global__ __launch_bounds__(256) void hproj_topk_kernel(
    const float* __restrict__ sp, const float* __restrict__ zxbcdt,
    const float* __restrict__ hw, const float* __restrict__ dt_bias,
    const float* __restrict__ gamma, const float* __restrict__ A_log,
    float* __restrict__ dt_ws, float* __restrict__ decay_ws) {
  __shared__ float s_udt[HIN];
  __shared__ float s_h[HOUT];
  const int r = blockIdx.x;
  const int tid = threadIdx.x;
  for (int i = tid; i < DM; i += 256) s_udt[i] = sp[(long)r * DM + i];
  if (tid < 32) s_udt[DM + tid] = zxbcdt[(long)r * NPROJ + (NPROJ - 32) + tid];
  __syncthreads();
  if (tid < HOUT) {
    const float* w = hw + tid * HIN;
    float acc = 0.f;
    for (int k = 0; k < HIN; ++k) acc += s_udt[k] * w[k];
    s_h[tid] = acc;
  }
  __syncthreads();
  if (tid == 0) {
    unsigned used = 0u;
    int ids[NH];
#pragma unroll 1
    for (int j = 0; j < NSEL; ++j) {          // selection top-k, ties -> low idx
      float best = -1e30f; int bi = 0;
      for (int i = 0; i < SELF; ++i)
        if (!((used >> i) & 1u) && s_h[i] > best) { best = s_h[i]; bi = i; }
      used |= 1u << bi;
      ids[j] = bi;
    }
    for (int j = NSEL; j < NH; ++j) ids[j] = SELF + (j - NSEL);  // shared 28..31
    const float dtb = dt_bias[0];
    for (int j = 0; j < NH; ++j) {
      float dv = s_udt[DM + ids[j]] + dtb;
      if (j < NSEL) dv += gamma[j] * s_h[SELF + j];   // spectral bias (zeros for shared)
      dv = dv > 20.f ? dv : log1pf(__expf(dv));       // softplus
      const float Aj = -__expf(A_log[j]);
      dt_ws[(long)r * NH + j] = dv;
      decay_ws[(long)r * NH + j] = __expf(dv * Aj);
    }
  }
}

// =====================================================================
// depthwise causal conv (width 4) + bias + SiLU over xBC channels
// =====================================================================
__global__ void conv_silu_kernel(const float* __restrict__ zxbcdt,
                                 const float* __restrict__ cw,
                                 const float* __restrict__ cb,
                                 float* __restrict__ xconv) {
  const int c = blockIdx.y * 256 + threadIdx.x;
  if (c >= CONVD) return;
  const int r = blockIdx.x;
  const int l = r & (L_ - 1);
  const float w0 = cw[c * 4 + 0], w1 = cw[c * 4 + 1];
  const float w2 = cw[c * 4 + 2], w3 = cw[c * 4 + 3];
  const float* xb = zxbcdt + (long)r * NPROJ + DSSM + c;  // xBC starts at col 1024
  float acc = cb[c];
  if (l >= 3) acc += xb[-3 * NPROJ] * w0;
  if (l >= 2) acc += xb[-2 * NPROJ] * w1;
  if (l >= 1) acc += xb[-1 * NPROJ] * w2;
  acc += xb[0] * w3;
  acc = acc / (1.f + __expf(-acc));                       // SiLU
  xconv[(long)r * CONVD + c] = acc;
}

// =====================================================================
// selective scan: block = (b, h, p-half). 256 thr: thread owns (p, 16 n's).
// state in VGPRs; B/C/x staged in LDS per step; n-reduction via shfl_xor
// within 8 consecutive lanes (wave32).  y += D[h]*x fused.
// =====================================================================
__global__ __launch_bounds__(256) void scan_kernel(
    const float* __restrict__ xconv, const float* __restrict__ dt_ws,
    const float* __restrict__ decay_ws, const float* __restrict__ Dv,
    float* __restrict__ y) {
  __shared__ float sB[DSTATE], sC[DSTATE], sX[32];
  const int blk = blockIdx.x;         // 0..127 = b*32 + h*2 + phalf
  const int b  = blk >> 5;
  const int h  = (blk >> 1) & 15;
  const int p0 = (blk & 1) * 32;
  const int tid = threadIdx.x;
  const int pl = tid >> 3;            // 0..31 -> local p
  const int nc = tid & 7;             // 8 n-chunks of 16
  const int n0 = nc * 16;
  float st[16];
#pragma unroll
  for (int i = 0; i < 16; ++i) st[i] = 0.f;
  const float Dh = Dv[h];
  const long rowbase = (long)b * L_;
  for (int t = 0; t < L_; ++t) {
    const long r = rowbase + t;
    const float* xc = xconv + r * CONVD;
    if (tid < DSTATE)            sB[tid]          = xc[DSSM + tid];
    else                         sC[tid - DSTATE] = xc[DSSM + DSTATE + (tid - DSTATE)];
    if (tid < 32)                sX[tid]          = xc[h * HD + p0 + tid];
    __syncthreads();
    const float dec = decay_ws[r * NH + h];
    const float dtt = dt_ws[r * NH + h];
    const float xv  = sX[pl];
    const float dtx = dtt * xv;
    float part = 0.f;
#pragma unroll
    for (int i = 0; i < 16; ++i) {
      st[i] = st[i] * dec + dtx * sB[n0 + i];
      part += st[i] * sC[n0 + i];
    }
    part += __shfl_xor(part, 1, 32);
    part += __shfl_xor(part, 2, 32);
    part += __shfl_xor(part, 4, 32);
    if (nc == 0) y[r * DSSM + h * HD + p0 + pl] = part + Dh * xv;
    __syncthreads();
  }
}

// =====================================================================
// yz = y * silu(z); RMS-norm over 1024; write bf16 for out_proj GEMM
// =====================================================================
__global__ __launch_bounds__(256) void rmsnorm_kernel(
    const float* __restrict__ y, const float* __restrict__ zxbcdt,
    const float* __restrict__ norm_w, __bf16* __restrict__ ybf) {
  __shared__ float red[256];
  const int r = blockIdx.x, tid = threadIdx.x;
  float yz[4]; float ss = 0.f;
#pragma unroll
  for (int j = 0; j < 4; ++j) {
    const int i = tid + j * 256;
    const float z = zxbcdt[(long)r * NPROJ + i];     // z = first 1024 cols
    const float v = y[(long)r * DSSM + i] * (z / (1.f + __expf(-z)));
    yz[j] = v; ss += v * v;
  }
  red[tid] = ss;
  __syncthreads();
  for (int s = 128; s > 0; s >>= 1) {
    if (tid < s) red[tid] += red[tid + s];
    __syncthreads();
  }
  const float rs = rsqrtf(red[0] / (float)DSSM + EPSV);
#pragma unroll
  for (int j = 0; j < 4; ++j) {
    const int i = tid + j * 256;
    ybf[(long)r * DSSM + i] = (__bf16)(yz[j] * rs * norm_w[i]);
  }
}

// =====================================================================
// launcher
// =====================================================================
extern "C" void kernel_launch(void* const* d_in, const int* in_sizes, int n_in,
                              void* d_out, int out_size, void* d_ws, size_t ws_size,
                              hipStream_t stream) {
  const float* u          = (const float*)d_in[0];
  const float* in_proj_w  = (const float*)d_in[1];
  const float* h_proj_w   = (const float*)d_in[2];
  const float* conv_w     = (const float*)d_in[3];
  const float* conv_b     = (const float*)d_in[4];
  const float* dt_bias    = (const float*)d_in[5];
  const float* gamma      = (const float*)d_in[6];
  const float* A_log      = (const float*)d_in[7];
  const float* Dvec       = (const float*)d_in[8];
  const float* norm_w     = (const float*)d_in[9];
  const float* out_proj_w = (const float*)d_in[10];
  float* out = (float*)d_out;

  // workspace carve-out (256B aligned)
  char* ws = (char*)d_ws;
  size_t off = 0;
  auto carve = [&](size_t bytes) -> char* {
    char* p = ws + off;
    off += (bytes + 255) & ~(size_t)255;
    return p;
  };
  float*  zxbcdt   = (float*) carve((size_t)BL * NPROJ * 4);
  float*  spectral = (float*) carve((size_t)BL * DM * 4);
  float*  xconv    = (float*) carve((size_t)BL * CONVD * 4);
  float*  ybuf     = (float*) carve((size_t)BL * DSSM * 4);
  float*  dt_ws    = (float*) carve((size_t)BL * NH * 4);
  float*  decay_ws = (float*) carve((size_t)BL * NH * 4);
  __bf16* u_bf     = (__bf16*)carve((size_t)BL * DM * 2);
  __bf16* w1_bf    = (__bf16*)carve((size_t)NPROJ * DM * 2);
  __bf16* w2_bf    = (__bf16*)carve((size_t)DM * DSSM * 2);
  __bf16* y_bf     = (__bf16*)carve((size_t)BL * DSSM * 2);

  // 1) bf16 staging for the tensor-core path
  cvt_bf16_kernel<<<2048, 256, 0, stream>>>(u, u_bf, BL * DM);
  cvt_bf16_kernel<<<2048, 256, 0, stream>>>(in_proj_w, w1_bf, NPROJ * DM);
  cvt_bf16_kernel<<<1024, 256, 0, stream>>>(out_proj_w, w2_bf, DM * DSSM);

  // 2) in_proj GEMM (8192 x 2336 x 1024) — WMMA bf16
  gemm_bf16_wmma<<<dim3(NPROJ / 32, BL / 128), 128, 0, stream>>>(
      u_bf, w1_bf, zxbcdt, BL, NPROJ, DM);

  // 3) spectral residual (causal prefix mean)
  spectral_kernel<<<B_ * 4, 256, 0, stream>>>(u, spectral);

  // 4) h_proj + top-k + dt/decay
  hproj_topk_kernel<<<BL, 256, 0, stream>>>(spectral, zxbcdt, h_proj_w,
                                            dt_bias, gamma, A_log,
                                            dt_ws, decay_ws);

  // 5) depthwise conv + SiLU
  conv_silu_kernel<<<dim3(BL, (CONVD + 255) / 256), 256, 0, stream>>>(
      zxbcdt, conv_w, conv_b, xconv);

  // 6) selective scan (+ D*x fused)
  scan_kernel<<<B_ * NH * 2, 256, 0, stream>>>(xconv, dt_ws, decay_ws,
                                               Dvec, ybuf);

  // 7) gated RMS-norm -> bf16
  rmsnorm_kernel<<<BL, 256, 0, stream>>>(ybuf, zxbcdt, norm_w, y_bf);

  // 8) out_proj GEMM (8192 x 1024 x 1024) — WMMA bf16
  gemm_bf16_wmma<<<dim3(DM / 32, BL / 128), 128, 0, stream>>>(
      y_bf, w2_bf, out, BL, DM, DSSM);
}
